// LSTM_35820027248832
// MI455X (gfx1250) — compile-verified, matrix operand order checked by reading, not run on hardware
//
#include <hip/hip_runtime.h>

// ---------------------------------------------------------------------------
// LSTM on MI455X (gfx1250):
//  - 64 persistent WGs (one per WGP), each owns 16 h-columns.
//  - Weights staged ONCE into 256 KB of LDS via the Tensor Data Mover
//    (tensor_load_to_lds + s_wait_tensorcnt), chunk-major layout; all 512
//    recurrent steps read B-fragments with ds_load_b128 (no L2 weight traffic).
//  - 8 waves/WG: waves 0-3 do x_t@w_ih^T, waves 4-7 do h_t@w_hh^T (K split),
//    partials combined through a small LDS exchange buffer.
//  - c in registers for the whole sequence; h ping-pong bf16 in global;
//    device-wide generation barrier per step.
// ---------------------------------------------------------------------------

#define T_STEPS  512
#define BATCH    64
#define HID      1024
#define NWG      64                  // one WG per WGP; each owns HID/NWG=16 cols
#define NTHREADS 256                 // 8 waves of 32 (wave32)

#define WMAT_BYTES   131072          // one weight slice: 64 rows x 1024 K x bf16
#define XCH_OFF      (2 * WMAT_BYTES)         // 262144
#define XCH_BYTES    16384                    // 4 mtiles x 4 gates x 32 lanes x 32B
#define SMEM_BYTES   (XCH_OFF + XCH_BYTES)    // 278528 <= 320KB LDS per WGP

typedef __attribute__((ext_vector_type(16))) __bf16 v16bf;
typedef __attribute__((ext_vector_type(8)))  float  v8f;

union FragBF16 { v16bf v; uint4 q[2]; };
union FragF32  { v8f   v; uint4 q[2]; };

__device__ __forceinline__ float sigmoid_f(float x) {
  return 1.0f / (1.0f + __expf(-x));
}
__device__ __forceinline__ float tanh_f(float x) {
  float e = __expf(2.0f * x);       // tanh = 1 - 2/(e^{2x}+1); safe at +/-inf
  return 1.0f - 2.0f / (e + 1.0f);
}

// Generation barrier across NWG persistent workgroups.
__device__ __forceinline__ void grid_sync(unsigned* bar, unsigned target) {
  __threadfence();
  __syncthreads();
  if (threadIdx.x == 0) {
    unsigned prev = __hip_atomic_fetch_add(&bar[0], 1u, __ATOMIC_ACQ_REL,
                                           __HIP_MEMORY_SCOPE_AGENT);
    if (prev == (unsigned)(NWG - 1)) {
      __hip_atomic_store(&bar[0], 0u, __ATOMIC_RELAXED, __HIP_MEMORY_SCOPE_AGENT);
      __hip_atomic_store(&bar[1], target, __ATOMIC_RELEASE, __HIP_MEMORY_SCOPE_AGENT);
    } else {
      while (__hip_atomic_load(&bar[1], __ATOMIC_ACQUIRE,
                               __HIP_MEMORY_SCOPE_AGENT) < target) {
        __builtin_amdgcn_s_sleep(2);
      }
    }
  }
  __syncthreads();
  __threadfence();
}

__global__ void cvt_f32_bf16(const float* __restrict__ in,
                             __bf16* __restrict__ out, int n) {
  int i = blockIdx.x * blockDim.x + threadIdx.x;
  int stride = gridDim.x * blockDim.x;
  for (; i < n; i += stride) out[i] = (__bf16)in[i];
}

#if defined(__has_builtin)
#if __has_builtin(__builtin_amdgcn_tensor_load_to_lds)
#define USE_TDM 1
#endif
#endif

#ifdef USE_TDM
typedef __attribute__((ext_vector_type(4))) unsigned int u32x4;
typedef __attribute__((ext_vector_type(8))) int          i32x8;
typedef __attribute__((ext_vector_type(4))) int          i32x4;

// TDM: copy one gate slice (16 rows x 1024 K, bf16, row stride 1024 elems)
// from global into LDS chunk-major: iteration i writes 16 rows x 8 elems
// (256 B contiguous) at lds_byte + i*1024B; global advances 8 elems per iter.
// This toolchain (clang-23/therock-10.0) uses the 6-arg builtin:
//   (u32x4 group0, i32x8 group1, i32x4 group2, i32x4 group3, i32x8 pad, i32 cpol)
// The ISA defines only four D# SGPR groups (VADDR4 unused -> zeros).
__device__ __forceinline__ void tdm_load_gate_slice(unsigned lds_byte,
                                                    unsigned long long ga) {
  u32x4 g0;
  g0[0] = 1u;                                            // count=1 (valid, user)
  g0[1] = lds_byte;                                      // lds_addr (bytes)
  g0[2] = (unsigned)(ga & 0xffffffffull);                // global_addr[31:0]
  g0[3] = (unsigned)((ga >> 32) & 0x01ffffffull) | (2u << 30); // [56:32]|type=2
  i32x8 g1;
  g1[0] = (int)((1u << 16) | (1u << 19));  // data_size=2B, iterate_enable=1
  g1[1] = (int)(1024u << 16);              // tensor_dim0 = 1024 (bits 79:48)
  g1[2] = (int)(16u << 16);                // tensor_dim1 = 16   (bits 111:80)
  g1[3] = (int)(8u << 16);                 // tile_dim0   = 8    (bits 127:112)
  g1[4] = 16;                              // tile_dim1=16, tile_dim2=0
  g1[5] = 1024;                            // tensor_dim0_stride = 1024 elems
  g1[6] = 0;
  g1[7] = 0;                               // tensor_dim1_stride unused (2D tile)
  i32x4 g2;
  g2[0] = 0;                               // tensor_dim2 (unused)
  g2[1] = 512;                             // lds_addr_increment = 1024B (elems)
  g2[2] = 8;                               // global_addr_increment = 8 elems
  g2[3] = (int)(127u << 16);               // iterate_count: 128 iterations
  i32x4 g3 = {0, 0, 0, 0};
  i32x8 g4 = {0, 0, 0, 0, 0, 0, 0, 0};     // VADDR4 unused per ISA
  __builtin_amdgcn_tensor_load_to_lds(g0, g1, g2, g3, g4, 0);
}
#endif

__global__ __launch_bounds__(NTHREADS, 1)
void lstm_persistent(const __bf16* __restrict__ xbf,    // [T, B, HID] bf16
                     const __bf16* __restrict__ wihbf,  // [4H, HID] bf16
                     const __bf16* __restrict__ whhbf,  // [4H, HID] bf16
                     const float*  __restrict__ b_ih,   // [4H]
                     const float*  __restrict__ b_hh,   // [4H]
                     const float*  __restrict__ c0,     // [B, HID]
                     __bf16*       __restrict__ hbuf,   // 2 x [B, HID] bf16
                     float*        __restrict__ out,    // h then c (fp32)
                     unsigned*     __restrict__ bar) {
  extern __shared__ __align__(16) char smem[];

  const int lane  = threadIdx.x & 31;
  const int wave  = threadIdx.x >> 5;
  const int grp   = wave >> 2;                // 0: x@w_ih part, 1: h@w_hh part
  const int mtile = wave & 3;                 // 4 row-tiles cover BATCH=64
  const int l15   = lane & 15;
  const int half  = lane >> 4;                // K-chunk selector (ISA A/B layout)
  const int kh8   = half * 8;

  const int col0  = blockIdx.x * 16;          // this WG's h-column base
  const int col   = col0 + l15;
  const int crow0 = mtile * 16 + half * 8;    // C/D row base (VGPR r -> row+r)
  const int arow  = mtile * 16 + l15;         // A-matrix row for this lane

  // ---- Stage both weight slices into LDS, chunk-major:
  //      byte addr = mat*128K + (k/8)*1024 + (gate*16 + localcol)*16
#ifdef USE_TDM
  if (wave == 0) {
#pragma unroll
    for (int mat = 0; mat < 2; ++mat) {
      const __bf16* wb = mat ? whhbf : wihbf;
#pragma unroll
      for (int g = 0; g < 4; ++g) {
        unsigned lb = (unsigned)(size_t)(void*)smem +
                      (unsigned)(mat * WMAT_BYTES + g * 256);
        unsigned long long ga =
            (unsigned long long)(size_t)(wb + (size_t)(g * HID + col0) * HID);
        tdm_load_gate_slice(lb, ga);
      }
    }
    __builtin_amdgcn_s_wait_tensorcnt(0);
  }
#else
  for (int mat = 0; mat < 2; ++mat) {
    const __bf16* wb = mat ? whhbf : wihbf;
    for (int idx = threadIdx.x; idx < 64 * 128; idx += NTHREADS) {
      int row = idx & 63, chunk = idx >> 6;
      int g = row >> 4, c = row & 15;
      const __bf16* src = wb + (size_t)(g * HID + col0 + c) * HID + chunk * 8;
      *(uint4*)&smem[mat * WMAT_BYTES + chunk * 1024 + row * 16] =
          *(const uint4*)src;
    }
  }
#endif
  __syncthreads();

  // Bias (grp 0 only folds it in; grp 1 starts at zero)
  v8f bias[4];
#pragma unroll
  for (int g = 0; g < 4; ++g) {
    float bb = b_ih[g * HID + col] + b_hh[g * HID + col];
#pragma unroll
    for (int r = 0; r < 8; ++r) bias[g][r] = bb;
  }

  v8f creg;
  if (grp == 0) {
#pragma unroll
    for (int r = 0; r < 8; ++r)
      creg[r] = c0[(size_t)(crow0 + r) * HID + col];
  }

  const char* matbase = smem + (grp ? WMAT_BYTES : 0);

  for (int t = 0; t < T_STEPS; ++t) {
    v8f acc[4];
#pragma unroll
    for (int g = 0; g < 4; ++g) {
      if (grp == 0) acc[g] = bias[g];
      else {
#pragma unroll
        for (int r = 0; r < 8; ++r) acc[g][r] = 0.0f;
      }
    }

    // This wave's K-half: grp0 reads x_t, grp1 reads h_t.
    const __bf16* A = (grp == 0) ? (xbf + (size_t)t * (BATCH * HID))
                                 : (hbuf + (size_t)(t & 1) * (BATCH * HID));
    const __bf16* aptr = A + (size_t)arow * HID;

#pragma unroll 2
    for (int k = 0; k < HID; k += 32) {
      FragBF16 fa;
      fa.q[0] = *(const uint4*)(aptr + k + kh8);
      fa.q[1] = *(const uint4*)(aptr + k + 16 + kh8);
      const char* cb = matbase + ((k >> 3) + half) * 1024;
#pragma unroll
      for (int g = 0; g < 4; ++g) {
        FragBF16 fb;
        const char* bp = cb + (g * 16 + l15) * 16;
        fb.q[0] = *(const uint4*)bp;            // K chunk (k/8 + half)
        fb.q[1] = *(const uint4*)(bp + 2048);   // K chunk (k/8 + 2 + half)
        acc[g] = __builtin_amdgcn_wmma_f32_16x16x32_bf16(
            false, fa.v, false, fb.v, (short)0, acc[g], false, false);
      }
    }

    // Combine the two K-halves through LDS, then gate math on grp 0.
    if (grp == 1) {
#pragma unroll
      for (int g = 0; g < 4; ++g) {
        FragF32 fx; fx.v = acc[g];
        uint4* dst = (uint4*)&smem[XCH_OFF + ((mtile * 4 + g) * 32 + lane) * 32];
        dst[0] = fx.q[0];
        dst[1] = fx.q[1];
      }
    }
    __syncthreads();

    if (grp == 0) {
      __bf16* hnext = hbuf + (size_t)((t + 1) & 1) * (BATCH * HID);
      const bool last = (t == T_STEPS - 1);
#pragma unroll
      for (int g = 0; g < 4; ++g) {
        const uint4* src =
            (const uint4*)&smem[XCH_OFF + ((mtile * 4 + g) * 32 + lane) * 32];
        FragF32 fp; fp.q[0] = src[0]; fp.q[1] = src[1];
        acc[g] = acc[g] + fp.v;
      }
#pragma unroll
      for (int r = 0; r < 8; ++r) {
        float ig = sigmoid_f(acc[0][r]);
        float fg = sigmoid_f(acc[1][r]);
        float gg = tanh_f(acc[2][r]);
        float og = sigmoid_f(acc[3][r]);
        float cn = fg * creg[r] + ig * gg;
        creg[r]  = cn;
        float hn = og * tanh_f(cn);
        const size_t idx = (size_t)(crow0 + r) * HID + col;
        hnext[idx] = (__bf16)hn;
        if (last) {
          out[idx] = hn;                        // h
          out[(size_t)BATCH * HID + idx] = cn;  // c
        }
      }
    }

    grid_sync(bar, (unsigned)(t + 1));
  }
}

extern "C" void kernel_launch(void* const* d_in, const int* in_sizes, int n_in,
                              void* d_out, int out_size, void* d_ws, size_t ws_size,
                              hipStream_t stream) {
  const float* x    = (const float*)d_in[0];
  const float* h0   = (const float*)d_in[1];
  const float* c0   = (const float*)d_in[2];
  const float* w_ih = (const float*)d_in[3];
  const float* w_hh = (const float*)d_in[4];
  const float* b_ih = (const float*)d_in[5];
  const float* b_hh = (const float*)d_in[6];
  float* out = (float*)d_out;

  char* ws = (char*)d_ws;
  unsigned* bar = (unsigned*)ws;                                   // 256 B
  size_t off = 256;
  __bf16* xbf   = (__bf16*)(ws + off); off += (size_t)T_STEPS * BATCH * HID * 2;
  __bf16* wihbf = (__bf16*)(ws + off); off += (size_t)4 * HID * HID * 2;
  __bf16* whhbf = (__bf16*)(ws + off); off += (size_t)4 * HID * HID * 2;
  __bf16* hbuf  = (__bf16*)(ws + off); off += (size_t)2 * BATCH * HID * 2;

  (void)hipFuncSetAttribute((const void*)lstm_persistent,
                            hipFuncAttributeMaxDynamicSharedMemorySize,
                            SMEM_BYTES);

  (void)hipMemsetAsync(bar, 0, 256, stream);

  cvt_f32_bf16<<<4096, NTHREADS, 0, stream>>>(x, xbf, T_STEPS * BATCH * HID);
  cvt_f32_bf16<<<2048, NTHREADS, 0, stream>>>(w_ih, wihbf, 4 * HID * HID);
  cvt_f32_bf16<<<2048, NTHREADS, 0, stream>>>(w_hh, whhbf, 4 * HID * HID);
  cvt_f32_bf16<<<256, NTHREADS, 0, stream>>>(h0, hbuf, BATCH * HID);

  lstm_persistent<<<NWG, NTHREADS, SMEM_BYTES, stream>>>(
      xbf, wihbf, whhbf, b_ih, b_hh, c0, hbuf, out, bar);
}
